// ConvEGNN3_17532056502806
// MI455X (gfx1250) — matrix operand order repaired and checked by reading, not compiled, not run on hardware
//
#include <hip/hip_runtime.h>
#include <hip/hip_bf16.h>
#include <math.h>

typedef __attribute__((ext_vector_type(16))) _Float16 v16h;
typedef __attribute__((ext_vector_type(4)))  _Float16 v4h;
typedef __attribute__((ext_vector_type(8)))  float    v8f;

#define K_NEIGH 64
#define IN_DIM  64
#define HID     64
#define LD      68   // padded LDS row stride (halfs) for 64-wide f16 tiles

__device__ __forceinline__ float silu_f(float x) {
  // x * sigmoid(x), sigmoid via v_exp_f32 + v_rcp_f32 (no IEEE divide sequence)
  return x * __builtin_amdgcn_rcpf(1.0f + __expf(-x));
}

// A fragment: 16x32 f16 tile, rows rowbase..rowbase+15 of row-major As (stride ld).
// ISA 7.12.2: lane L holds row L%16; sel=L/16; half j<8 -> K=k0+sel*8+j, j>=8 -> K=k0+16+sel*8+(j-8)
__device__ __forceinline__ v16h load_a_frag(const _Float16* As, int ld, int rowbase, int k0, int lane) {
  const _Float16* p = As + (rowbase + (lane & 15)) * ld + k0 + ((lane >> 4) & 1) * 8;
  v16h a;
#pragma unroll
  for (int j = 0; j < 8; ++j) a[j] = p[j];
#pragma unroll
  for (int j = 0; j < 8; ++j) a[j + 8] = p[16 + j];
  return a;
}

// B fragment: 32x16 f16 from TRANSPOSED weight tile Wt[n][k] (stride ld).
// ISA: lane L holds column N=L%16; sel=L/16; half j<8 -> K=k0+sel*8+j, j>=8 -> K=k0+16+sel*8+(j-8)
__device__ __forceinline__ v16h load_b_fragT(const _Float16* Wt, int ld, int k0, int col0, int lane) {
  const _Float16* p = Wt + (col0 + (lane & 15)) * ld + k0 + ((lane >> 4) & 1) * 8;
  v16h b;
#pragma unroll
  for (int j = 0; j < 8; ++j) b[j] = p[j];
#pragma unroll
  for (int j = 0; j < 8; ++j) b[j + 8] = p[16 + j];
  return b;
}

__device__ __forceinline__ v4h cvt4(const float4 f) {
  v4h h;
  h[0] = (_Float16)f.x; h[1] = (_Float16)f.y;
  h[2] = (_Float16)f.z; h[3] = (_Float16)f.w;
  return h;
}

__global__ __launch_bounds__(128)
void egnn_node_kernel(const float* __restrict__ ped, const float* __restrict__ hst,
                      const float* __restrict__ hneigh, const float* __restrict__ rela,
                      const int* __restrict__ mask,
                      const float* __restrict__ We1, const float* __restrict__ be1,
                      const float* __restrict__ We2, const float* __restrict__ be2,
                      const float* __restrict__ Wx1, const float* __restrict__ bx1,
                      const float* __restrict__ Wx2, const float* __restrict__ bx2,
                      const float* __restrict__ Wh1, const float* __restrict__ bh1,
                      const float* __restrict__ Wh2, const float* __restrict__ bh2,
                      float* __restrict__ out_x, float* __restrict__ out_h)
{
  __shared__ _Float16 sA  [K_NEIGH * LD];   // h_neigh tile, edges x 64 (f16)
  __shared__ _Float16 sW1t[HID * LD];       // We1 rows 64..127 transposed [n][k]
  __shared__ _Float16 sW2t[HID * LD];       // We2 transposed
  __shared__ _Float16 sWx1t[HID * LD];      // Wx1 transposed
  __shared__ _Float16 sH  [K_NEIGH * LD];   // hidden1 / gate-hidden
  __shared__ _Float16 sM  [K_NEIGH * LD];   // masked m_ij (f16)
  __shared__ float sBase[HID];              // be1 + h_st @ We1[0:64]  (broadcast part)
  __shared__ float sWl[HID];                // We1 row 128 (dist column)
  __shared__ float sHst[IN_DIM], sDist[K_NEIGH], sRX[K_NEIGH], sRY[K_NEIGH];
  __shared__ float sB2[HID], sBx1[HID], sWx2f[HID];
  __shared__ float sGate[K_NEIGH], sMi[HID], sHid[HID];
  __shared__ int   sMk[K_NEIGH];

  const int tid  = threadIdx.x;
  const int lane = tid & 31;
  const int wave = tid >> 5;
  const int bn   = blockIdx.x;

  const float* hst_g = hst + (size_t)bn * IN_DIM;
  const float* hn_g  = hneigh + (size_t)bn * K_NEIGH * IN_DIM;
  const float* rl_g  = rela + (size_t)bn * K_NEIGH * 6;
  const int*   mk_g  = mask + (size_t)bn * K_NEIGH;

  // ---------------- Phase 0: stage into LDS (f32 -> f16, vectorized) ----------------
  // h_neigh: 64x64, float4 loads, 4-half LDS stores
  for (int g = tid; g < K_NEIGH * (IN_DIM / 4); g += 128) {
    int r = g >> 4, c4 = (g & 15) * 4;
    float4 f = *(const float4*)(hn_g + r * IN_DIM + c4);
    *(v4h*)(sA + r * LD + c4) = cvt4(f);
  }
  // weights, transposed: each group handles 4 consecutive k for one column c
  for (int g = tid; g < HID * (HID / 4); g += 128) {
    int c = g & 63, k0 = (g >> 6) * 4;
    {
      float4 f;
      f.x = We1[(64 + k0 + 0) * HID + c]; f.y = We1[(64 + k0 + 1) * HID + c];
      f.z = We1[(64 + k0 + 2) * HID + c]; f.w = We1[(64 + k0 + 3) * HID + c];
      *(v4h*)(sW1t + c * LD + k0) = cvt4(f);
    }
    {
      float4 f;
      f.x = We2[(k0 + 0) * HID + c]; f.y = We2[(k0 + 1) * HID + c];
      f.z = We2[(k0 + 2) * HID + c]; f.w = We2[(k0 + 3) * HID + c];
      *(v4h*)(sW2t + c * LD + k0) = cvt4(f);
    }
    {
      float4 f;
      f.x = Wx1[(k0 + 0) * HID + c]; f.y = Wx1[(k0 + 1) * HID + c];
      f.z = Wx1[(k0 + 2) * HID + c]; f.w = Wx1[(k0 + 3) * HID + c];
      *(v4h*)(sWx1t + c * LD + k0) = cvt4(f);
    }
  }
  if (tid < 64) {
    // broadcast part of layer 1: base[n] = be1[n] + sum_j h_st[j]*We1[j][n]
    float a = be1[tid];
#pragma unroll 4
    for (int j = 0; j < IN_DIM; ++j) a = fmaf(hst_g[j], We1[j * HID + tid], a);
    sBase[tid] = a;
    sWl[tid]   = We1[128 * HID + tid];
    sB2[tid]   = be2[tid];
    sBx1[tid]  = bx1[tid];
    sWx2f[tid] = Wx2[tid];
    sHst[tid]  = hst_g[tid];
    float rx = rl_g[tid * 6 + 0];
    float ry = rl_g[tid * 6 + 1];
    sRX[tid] = rx; sRY[tid] = ry;
    sDist[tid] = sqrtf(rx * rx + ry * ry);
    sMk[tid] = mk_g[tid];
  }
  __syncthreads();

  const int rowbase = wave * 16;   // each wave owns 16 edges
  const int col = lane & 15;
  const int sel = (lane >> 4) & 1;

  // ---------------- GEMM1: h_neigh(64x64) @ We1[64:128] + base + dist rank-1 ----------------
#pragma unroll
  for (int nt = 0; nt < 4; ++nt) {
    v8f acc = {0.f, 0.f, 0.f, 0.f, 0.f, 0.f, 0.f, 0.f};
#pragma unroll
    for (int kt = 0; kt < 2; ++kt) {
      v16h a = load_a_frag(sA, LD, rowbase, kt * 32, lane);
      v16h b = load_b_fragT(sW1t, LD, kt * 32, nt * 16, lane);
      acc = __builtin_amdgcn_wmma_f32_16x16x32_f16(false, a, false, b, (short)0, acc, false, false);
    }
    const int n = nt * 16 + col;
    const float wl = sWl[n], bb = sBase[n];
#pragma unroll
    for (int v = 0; v < 8; ++v) {
      int m = rowbase + sel * 8 + v;
      float x = fmaf(sDist[m], wl, acc[v] + bb);
      sH[m * LD + n] = (_Float16)silu_f(x);
    }
  }
  __syncthreads();

  // ---------------- GEMM2: hidden1 @ We2, SiLU, mask -> m_ij ----------------
#pragma unroll
  for (int nt = 0; nt < 4; ++nt) {
    v8f acc = {0.f, 0.f, 0.f, 0.f, 0.f, 0.f, 0.f, 0.f};
#pragma unroll
    for (int kt = 0; kt < 2; ++kt) {
      v16h a = load_a_frag(sH, LD, rowbase, kt * 32, lane);
      v16h b = load_b_fragT(sW2t, LD, kt * 32, nt * 16, lane);
      acc = __builtin_amdgcn_wmma_f32_16x16x32_f16(false, a, false, b, (short)0, acc, false, false);
    }
    const int n = nt * 16 + col;
    const float bb = sB2[n];
#pragma unroll
    for (int v = 0; v < 8; ++v) {
      int m = rowbase + sel * 8 + v;
      float x = silu_f(acc[v] + bb);
      sM[m * LD + n] = (_Float16)(sMk[m] ? x : 0.0f);
    }
  }
  __syncthreads();

  // ---------------- GEMM3: m_ij @ Wx1, SiLU -> gate hidden (reuse sH) ----------------
#pragma unroll
  for (int nt = 0; nt < 4; ++nt) {
    v8f acc = {0.f, 0.f, 0.f, 0.f, 0.f, 0.f, 0.f, 0.f};
#pragma unroll
    for (int kt = 0; kt < 2; ++kt) {
      v16h a = load_a_frag(sM, LD, rowbase, kt * 32, lane);
      v16h b = load_b_fragT(sWx1t, LD, kt * 32, nt * 16, lane);
      acc = __builtin_amdgcn_wmma_f32_16x16x32_f16(false, a, false, b, (short)0, acc, false, false);
    }
    const int n = nt * 16 + col;
    const float bb = sBx1[n];
#pragma unroll
    for (int v = 0; v < 8; ++v) {
      int m = rowbase + sel * 8 + v;
      sH[m * LD + n] = (_Float16)silu_f(acc[v] + bb);
    }
  }
  __syncthreads();

  // ---------------- gate scalar (lower half) + m_i column sums (upper half) ----------------
  if (tid < 64) {
    float g = bx2[0];
#pragma unroll 8
    for (int j = 0; j < HID; ++j) g = fmaf((float)sH[tid * LD + j], sWx2f[j], g);
    sGate[tid] = g;
  } else {
    int t = tid - 64;
    float s = 0.f;
#pragma unroll 8
    for (int k = 0; k < K_NEIGH; ++k) s += (float)sM[k * LD + t];
    sMi[t] = s;
  }
  __syncthreads();

  // ---------------- x_new (threads 64,65) || f_h hidden (threads 0..63) ----------------
  if (tid >= 64 && tid < 66) {
    int t = tid - 64;
    const float* rv = t ? sRY : sRX;
    float s = 0.f; int c = 0;
#pragma unroll 8
    for (int k = 0; k < K_NEIGH; ++k) { s = fmaf(rv[k], sGate[k], s); c += (sMk[k] != 0); }
    out_x[bn * 2 + t] = ped[bn * 2 + t] + s * __builtin_amdgcn_rcpf((float)c + 1e-6f);
  }
  if (tid < 64) {
    float a1 = bh1[tid];
#pragma unroll 4
    for (int j = 0; j < IN_DIM; ++j) a1 = fmaf(sHst[j], Wh1[j * HID + tid], a1);        // coalesced
#pragma unroll 4
    for (int j = 0; j < HID; ++j) a1 = fmaf(sMi[j], Wh1[(IN_DIM + j) * HID + tid], a1); // coalesced
    sHid[tid] = silu_f(a1);
  }
  __syncthreads();
  if (tid < 64) {
    float a2 = bh2[tid];
#pragma unroll 4
    for (int j = 0; j < HID; ++j) a2 = fmaf(sHid[j], Wh2[j * HID + tid], a2);           // coalesced
    out_h[(size_t)bn * HID + tid] = sHst[tid] + a2;
  }
}

extern "C" void kernel_launch(void* const* d_in, const int* in_sizes, int n_in,
                              void* d_out, int out_size, void* d_ws, size_t ws_size,
                              hipStream_t stream) {
  const float* ped = (const float*)d_in[0];
  const float* hst = (const float*)d_in[1];
  const float* hn  = (const float*)d_in[2];
  const float* rl  = (const float*)d_in[3];
  const int*   mk  = (const int*)d_in[4];
  const float* We1 = (const float*)d_in[5];
  const float* be1 = (const float*)d_in[6];
  const float* We2 = (const float*)d_in[7];
  const float* be2 = (const float*)d_in[8];
  const float* Wx1 = (const float*)d_in[9];
  const float* bx1 = (const float*)d_in[10];
  const float* Wx2 = (const float*)d_in[11];
  const float* bx2 = (const float*)d_in[12];
  const float* Wh1 = (const float*)d_in[13];
  const float* bh1 = (const float*)d_in[14];
  const float* Wh2 = (const float*)d_in[15];
  const float* bh2 = (const float*)d_in[16];

  const int nodes = in_sizes[1] / IN_DIM;   // BS * N = 16384
  float* out_x = (float*)d_out;                       // [nodes, 2]
  float* out_h = (float*)d_out + (size_t)nodes * 2;   // [nodes, HID]

  egnn_node_kernel<<<dim3(nodes), dim3(128), 0, stream>>>(
      ped, hst, hn, rl, mk, We1, be1, We2, be2, Wx1, bx1, Wx2, bx2,
      Wh1, bh1, Wh2, bh2, out_x, out_h);
}
